// mmTEM_31997506355366
// MI455X (gfx1250) — matrix operand order repaired and checked by reading, not compiled, not run on hardware
//
#include <hip/hip_runtime.h>
#include <hip/hip_bf16.h>

typedef __attribute__((ext_vector_type(16))) _Float16 v16h;
typedef __attribute__((ext_vector_type(8)))  _Float16 v8h;
typedef __attribute__((ext_vector_type(4)))  _Float16 v4h;
typedef __attribute__((ext_vector_type(8)))  float    v8f;
typedef __attribute__((ext_vector_type(4)))  float    v4f;

#define FLAG_ACCUM  1
#define FLAG_BIAS   2
#define FLAG_RELU   4
#define FLAG_TRANSB 8
#define FLAG_MASK   16

// W layout constants (meta-MLP flattened param vector)
#define W_TOT   131712
#define OFF_W1  0
#define OFF_B1  32768
#define OFF_W2  33024
#define OFF_B2  98560
#define OFF_W3  98816
#define OFF_B3  131584
#define STATE_PER_B (2 * W_TOT)   // 263424

__device__ __forceinline__ float sigm(float x) { return 1.0f / (1.0f + expf(-x)); }

// ---------------------------------------------------------------------------
// Generic fp32-in/fp32-out GEMM via v_wmma_f32_16x16x32_f16.
// Block: 256 threads = 8 waves, tile BM=32 x BN=64, BK=32.
// Requires M%32==0, N%64==0, K%32==0, 16B-aligned rows (true at all call sites).
// LDS: A row-major [32][40], B transposed [n][k] so both fragments are
// contiguous -> ds_load_b128 pairs. Staging uses global_load_b128.
// ---------------------------------------------------------------------------
__global__ __launch_bounds__(256) void gemm_k(
    const float* __restrict__ A, int lda,
    const float* __restrict__ B, int ldb,
    const float* __restrict__ bias,
    const float* __restrict__ aux,
    float* __restrict__ C,
    int M, int N, int K, int flags)
{
  __shared__ _Float16 As[32][40];    // [m][k], row = 80B (16B multiple)
  __shared__ _Float16 BsT[64][40];   // [n][k]
  int tid = threadIdx.x;
  int m0 = blockIdx.y * 32;
  int n0 = blockIdx.x * 64;
  int wave = tid >> 5;
  int lane = tid & 31;
  int wm = wave >> 2;         // 0..1
  int wn = wave & 3;          // 0..3
  int hlf = lane >> 4;        // 0/1
  int r16 = lane & 15;
  int koffA = hlf * 8;

  int ar  = tid >> 3;         // 0..31  (A staging row)
  int ac4 = (tid & 7) * 4;    // 0..28  (A staging col group)

  v8f acc = {0.f, 0.f, 0.f, 0.f, 0.f, 0.f, 0.f, 0.f};

  for (int kk = 0; kk < K; kk += 32) {
    // ---- stage A tile 32x32: one float4 per thread ----
    {
      v4f av = *(const v4f*)(A + (m0 + ar) * lda + kk + ac4);
      v4h ah;
      ah[0] = (_Float16)av[0]; ah[1] = (_Float16)av[1];
      ah[2] = (_Float16)av[2]; ah[3] = (_Float16)av[3];
      *(v4h*)&As[ar][ac4] = ah;
    }
    // ---- stage B tile 32x64 into transposed LDS ----
    if (flags & FLAG_TRANSB) {
      // B stored [N,K]: tile column c is contiguous in global -> vector loads
      #pragma unroll
      for (int l = 0; l < 2; ++l) {
        int f4 = tid + l * 256;
        int c  = f4 >> 3;          // 0..63
        int r4 = (f4 & 7) * 4;     // 0..28 (k group)
        v4f bv = *(const v4f*)(B + (n0 + c) * ldb + kk + r4);
        v4h bh;
        bh[0] = (_Float16)bv[0]; bh[1] = (_Float16)bv[1];
        bh[2] = (_Float16)bv[2]; bh[3] = (_Float16)bv[3];
        *(v4h*)&BsT[c][r4] = bh;
      }
    } else {
      // B stored [K,N]: vector load along n, scatter 4 halfs into BsT columns
      #pragma unroll
      for (int l = 0; l < 2; ++l) {
        int f4 = tid + l * 256;
        int r  = f4 >> 4;          // 0..31 (k)
        int c4 = (f4 & 15) * 4;    // 0..60 (n group)
        v4f bv = *(const v4f*)(B + (kk + r) * ldb + n0 + c4);
        BsT[c4 + 0][r] = (_Float16)bv[0];
        BsT[c4 + 1][r] = (_Float16)bv[1];
        BsT[c4 + 2][r] = (_Float16)bv[2];
        BsT[c4 + 3][r] = (_Float16)bv[3];
      }
    }
    __syncthreads();

    // ---- fragments: two aligned b128 LDS loads each ----
    v8h alo = *(const v8h*)&As[wm * 16 + r16][koffA];
    v8h ahi = *(const v8h*)&As[wm * 16 + r16][16 + koffA];
    v8h blo = *(const v8h*)&BsT[wn * 16 + r16][koffA];
    v8h bhi = *(const v8h*)&BsT[wn * 16 + r16][16 + koffA];
    v16h af = __builtin_shufflevector(alo, ahi, 0,1,2,3,4,5,6,7,8,9,10,11,12,13,14,15);
    v16h bf = __builtin_shufflevector(blo, bhi, 0,1,2,3,4,5,6,7,8,9,10,11,12,13,14,15);
    acc = __builtin_amdgcn_wmma_f32_16x16x32_f16(false, af, false, bf,
                                                 (short)0, acc, false, false);
    __syncthreads();
  }

  int gm_base = m0 + wm * 16 + hlf * 8;
  int gn = n0 + wn * 16 + r16;
  #pragma unroll
  for (int r = 0; r < 8; ++r) {
    int gm = gm_base + r;
    int ci = gm * N + gn;
    float v = acc[r];
    if (flags & FLAG_ACCUM) v += C[ci];
    if (flags & FLAG_BIAS)  v += bias[gn];
    if (flags & FLAG_RELU)  v = fmaxf(v, 0.0f);
    if (flags & FLAG_MASK)  v = (aux[ci] > 0.0f) ? v : 0.0f;
    C[ci] = v;
  }
}

// ---------------------------------------------------------------------------
// TTT final-state GEMM: both scan outputs collapse to rank-32 weighted sums.
//   upd[m,n] = sum_k wu[k] * A[k,m] * B[k,n]
//   acc[m,n] = Fprod * P[m,n] - sum_k wa[k] * A[k,m] * B[k,n]
// One wave per 16x16 tile, K=32 -> exactly two WMMAs per block.
// Per-element global loads are coalesced across lanes (consecutive m / n).
// ---------------------------------------------------------------------------
__global__ __launch_bounds__(32) void ttt_gemm_k(
    const float* __restrict__ Abase,   // [B,32,M]
    const float* __restrict__ Bbase,   // [B,32,N]
    const float* __restrict__ P,       // [M,N] shared param
    const float* __restrict__ wu, const float* __restrict__ wa,
    const float* __restrict__ fp,
    float* __restrict__ out,           // d_out + 1
    int M, int N, int accOff)
{
  int b = blockIdx.z;
  const float* A  = Abase + b * 32 * M;
  const float* Bm = Bbase + b * 32 * N;
  int m0 = blockIdx.y * 16;
  int n0 = blockIdx.x * 16;
  int lane = threadIdx.x;
  int hlf = lane >> 4;
  int r16 = lane & 15;
  int koffA = hlf * 8;

  __shared__ float wus[32], was[32];
  wus[lane] = wu[b * 32 + lane];
  was[lane] = wa[b * 32 + lane];
  __syncthreads();

  v16h af, bu, ba;
  #pragma unroll
  for (int e = 0; e < 8; ++e) {
    af[e]     = (_Float16)A[(koffA + e) * M + m0 + r16];
    af[e + 8] = (_Float16)A[(16 + koffA + e) * M + m0 + r16];
  }
  #pragma unroll
  for (int e = 0; e < 16; ++e) {
    int k = (e < 8) ? (hlf * 8 + e) : (16 + hlf * 8 + (e - 8));
    float v = Bm[k * N + n0 + r16];
    bu[e] = (_Float16)(wus[k] * v);
    ba[e] = (_Float16)(was[k] * v);
  }
  v8f z = {0.f, 0.f, 0.f, 0.f, 0.f, 0.f, 0.f, 0.f};
  v8f aU = __builtin_amdgcn_wmma_f32_16x16x32_f16(false, af, false, bu, (short)0, z, false, false);
  v8f aA = __builtin_amdgcn_wmma_f32_16x16x32_f16(false, af, false, ba, (short)0, z, false, false);

  float F = fp[b];
  float* obat = out + b * STATE_PER_B;
  #pragma unroll
  for (int r = 0; r < 8; ++r) {
    int gm = m0 + hlf * 8 + r;
    int gn = n0 + r16;
    int idx = gm * N + gn;
    obat[accOff + idx]         = F * P[idx] - aA[r];
    obat[W_TOT + accOff + idx] = aU[r];
  }
}

// TTT bias parts of the scan outputs
__global__ void ttt_bias_k(
    const float* __restrict__ dh1m, const float* __restrict__ dh2m,
    const float* __restrict__ dob,
    const float* __restrict__ m_b1, const float* __restrict__ m_b2,
    const float* __restrict__ m_b3,
    const float* __restrict__ wu, const float* __restrict__ wa,
    const float* __restrict__ fp, float* __restrict__ out)
{
  int n = blockIdx.x * blockDim.x + threadIdx.x;
  int ty = blockIdx.y, b = blockIdx.z;
  int len = (ty == 2) ? 128 : 256;
  if (n >= len) return;
  const float* Bv = (ty == 0) ? dh1m : (ty == 1) ? dh2m : dob;
  const float* pb = (ty == 0) ? m_b1 : (ty == 1) ? m_b2 : m_b3;
  int off = (ty == 0) ? OFF_B1 : (ty == 1) ? OFF_B2 : OFF_B3;
  float su = 0.f, sa = 0.f;
  for (int t = 0; t < 32; ++t) {
    float v = Bv[(b * 32 + t) * len + n];
    su += wu[b * 32 + t] * v;
    sa += wa[b * 32 + t] * v;
  }
  float* obat = out + b * STATE_PER_B;
  obat[off + n]         = fp[b] * pb[n] - sa;
  obat[W_TOT + off + n] = su;
}

// ---------------------------------------------------------------------------
// Sequential RNN scan: h = l2norm(relu(h @ Tm[b,t])), state in LDS.
// ---------------------------------------------------------------------------
__global__ __launch_bounds__(256) void rnn_scan_k(
    const float* __restrict__ trans, const float* __restrict__ init,
    float* __restrict__ structural)
{
  __shared__ float h[8][128];
  __shared__ float hn[8][128];
  __shared__ float nrm[8];
  int tid = threadIdx.x;
  if (tid == 0) {
    float s = 0.f;
    for (int i = 0; i < 128; ++i) s += init[i] * init[i];
    nrm[0] = fmaxf(sqrtf(s), 1e-12f);
  }
  __syncthreads();
  for (int rr = 0; rr < 4; ++rr) {
    int idx = tid + 256 * rr;
    int b = idx >> 7, j = idx & 127;
    h[b][j] = init[j] / nrm[0];
  }
  __syncthreads();
  for (int t = 0; t < 32; ++t) {
    for (int rr = 0; rr < 4; ++rr) {
      int idx = tid + 256 * rr;
      int b = idx >> 7, j = idx & 127;
      const float* Tm = trans + (b * 32 + t) * 128 * 128;
      float s = 0.f;
      #pragma unroll 8
      for (int i = 0; i < 128; ++i) s += h[b][i] * Tm[i * 128 + j];
      hn[b][j] = fmaxf(s, 0.f);
    }
    __syncthreads();
    if (tid < 8) {
      float s = 0.f;
      for (int j = 0; j < 128; ++j) s += hn[tid][j] * hn[tid][j];
      nrm[tid] = fmaxf(sqrtf(s), 1e-12f);
    }
    __syncthreads();
    for (int rr = 0; rr < 4; ++rr) {
      int idx = tid + 256 * rr;
      int b = idx >> 7, j = idx & 127;
      float v = hn[b][j] / nrm[b];
      h[b][j] = v;
      structural[(b * 32 + t) * 128 + j] = v;
    }
    __syncthreads();
  }
}

// ---------------------------------------------------------------------------
// Small helper kernels
// ---------------------------------------------------------------------------
__global__ void zero_loss_k(float* out) {
  if (blockIdx.x == 0 && threadIdx.x == 0) out[0] = 0.f;
}

// N must be a power of two (128/512 here): nSh = log2(N)
__global__ void mse_acc_k(const float* __restrict__ a, int lda,
                          const float* __restrict__ b, int ldb,
                          int M, int nSh, float scale, float* out)
{
  __shared__ float red[256];
  int tid = threadIdx.x;
  int total = M << nSh;
  int nMask = (1 << nSh) - 1;
  float s = 0.f;
  for (int idx = blockIdx.x * 256 + tid; idx < total; idx += gridDim.x * 256) {
    int m = idx >> nSh, n = idx & nMask;
    float d = a[m * lda + n] - b[m * ldb + n];
    s += d * d;
  }
  red[tid] = s;
  __syncthreads();
  for (int st = 128; st > 0; st >>= 1) {
    if (tid < st) red[tid] += red[tid + st];
    __syncthreads();
  }
  if (tid == 0) atomicAdd(out, red[0] * scale);
}

__global__ void sse_k(const float* __restrict__ d3, const float* __restrict__ enc,
                      float* __restrict__ sse)
{
  int m = blockIdx.x * blockDim.x + threadIdx.x;
  if (m >= 256) return;
  float s = 0.f;
  for (int n = 0; n < 128; ++n) {
    float d = d3[m * 256 + 128 + n] - enc[m * 128 + n];
    s += d * d;
  }
  sse[m] = s;
}

__global__ void vh_epi_k(float* __restrict__ vh, const float* __restrict__ sse,
                         const float* __restrict__ w_last, const float* __restrict__ b1)
{
  int i = blockIdx.x * blockDim.x + threadIdx.x;
  if (i >= 256 * 256) return;
  int m = i >> 8, n = i & 255;
  vh[i] = fmaxf(vh[i] + sse[m] * w_last[n] + b1[n], 0.f);
}

__global__ void infs_k(const float* __restrict__ d1, const float* __restrict__ d3,
                       const float* __restrict__ pvar, const float* __restrict__ ir,
                       float* __restrict__ infs)
{
  int i = blockIdx.x * blockDim.x + threadIdx.x;
  if (i >= 256 * 128) return;
  int m = i >> 7, n = i & 127;
  float dgs = d1[m * 256 + n];
  float cs  = d3[m * 256 + n];
  infs[i] = dgs + (cs - dgs) * sigm(ir[0]) * pvar[i];
}

__global__ void joint_k(const float* __restrict__ d4, const float* __restrict__ enc,
                        float* __restrict__ joint)
{
  int i = blockIdx.x * blockDim.x + threadIdx.x;
  if (i >= 256 * 256) return;
  int m = i >> 8, n = i & 255;
  joint[i] = (n < 128) ? d4[m * 256 + n] : enc[m * 128 + n - 128];
}

__global__ void ew_dout_k(const float* __restrict__ mo, const float* __restrict__ va,
                          float* __restrict__ dob, int n)
{
  int i = blockIdx.x * blockDim.x + threadIdx.x;
  if (i < n) dob[i] = (2.0f / 128.0f) * (mo[i] - va[i]);
}

__global__ void hp_k(const float* __restrict__ joint, const float* __restrict__ w_hp,
                     float* __restrict__ hp)
{
  int m = blockIdx.x * blockDim.x + threadIdx.x;
  if (m >= 256) return;
  float s0 = 0.f, s1 = 0.f, s2 = 0.f;
  for (int k = 0; k < 256; ++k) {
    float j = joint[m * 256 + k];
    s0 += j * w_hp[k * 3 + 0];
    s1 += j * w_hp[k * 3 + 1];
    s2 += j * w_hp[k * 3 + 2];
  }
  hp[m * 3 + 0] = s0; hp[m * 3 + 1] = s1; hp[m * 3 + 2] = s2;
}

// per-batch scalar gate chains: wu_r = lr_r * prod_{s>r} beta_s
// wa_r = lr_r * S_r with S_31=1, S_r = cf_r + beta_{r+1} S_{r+1}; Fprod = prod f
__global__ void coeff_k(const float* __restrict__ hp, float* __restrict__ wu,
                        float* __restrict__ wa, float* __restrict__ fp)
{
  int b = threadIdx.x;
  if (b >= 8) return;
  float lr[32], f[32], be[32];
  for (int t = 0; t < 32; ++t) {
    const float* h = hp + (b * 32 + t) * 3;
    lr[t] = h[0];
    f[t]  = sigm(h[1]);
    be[t] = sigm(h[2]);
  }
  float cf = 1.f, cb = 1.f, S = 1.f;
  wu[b * 32 + 31] = lr[31];
  wa[b * 32 + 31] = lr[31];
  for (int r = 30; r >= 0; --r) {
    cf *= f[r + 1];
    cb *= be[r + 1];
    S = cf + be[r + 1] * S;
    wu[b * 32 + r] = lr[r] * cb;
    wa[b * 32 + r] = lr[r] * S;
  }
  fp[b] = f[0] * cf;
}

// ---------------------------------------------------------------------------
static void launch_gemm(hipStream_t s, const float* A, int lda, const float* B, int ldb,
                        const float* bias, const float* aux, float* C,
                        int M, int N, int K, int flags)
{
  dim3 g(N / 64, M / 32);
  gemm_k<<<g, 256, 0, s>>>(A, lda, B, ldb, bias, aux, C, M, N, K, flags);
}

extern "C" void kernel_launch(void* const* d_in, const int* in_sizes, int n_in,
                              void* d_out, int out_size, void* d_ws, size_t ws_size,
                              hipStream_t stream)
{
  const float* sensory = (const float*)d_in[0];
  const float* actions = (const float*)d_in[1];
  const float* enc_w   = (const float*)d_in[2];
  const float* dec_w   = (const float*)d_in[3];
  const float* pi_w1   = (const float*)d_in[4];
  const float* pi_b1   = (const float*)d_in[5];
  const float* pi_w2   = (const float*)d_in[6];
  const float* pi_b2   = (const float*)d_in[7];
  const float* pi_w3   = (const float*)d_in[8];
  const float* pi_b3   = (const float*)d_in[9];
  const float* rnn_init = (const float*)d_in[10];
  const float* wq      = (const float*)d_in[11];
  const float* wk      = (const float*)d_in[12];
  const float* wv      = (const float*)d_in[13];
  const float* w_hp    = (const float*)d_in[14];
  const float* m_w1    = (const float*)d_in[15];
  const float* m_b1    = (const float*)d_in[16];
  const float* m_w2    = (const float*)d_in[17];
  const float* m_b2    = (const float*)d_in[18];
  const float* m_w3    = (const float*)d_in[19];
  const float* m_b3    = (const float*)d_in[20];
  const float* o_w1    = (const float*)d_in[21];
  const float* o_b1    = (const float*)d_in[22];
  const float* o_w2    = (const float*)d_in[23];
  const float* o_b2    = (const float*)d_in[24];
  const float* o_w3    = (const float*)d_in[25];
  const float* o_b3    = (const float*)d_in[26];
  const float* v_w1    = (const float*)d_in[27];
  const float* v_b1    = (const float*)d_in[28];
  const float* v_w2    = (const float*)d_in[29];
  const float* v_b2    = (const float*)d_in[30];
  const float* v_w3    = (const float*)d_in[31];
  const float* v_b3    = (const float*)d_in[32];
  const float* ir      = (const float*)d_in[33];

  float* out = (float*)d_out;
  float* Wsp = (float*)d_ws;
  size_t off = 0;
  auto alloc = [&](size_t n) { float* p = Wsp + off; off += n; return p; };

  float* trans = alloc((size_t)8 * 32 * 128 * 128);
  float* h1p  = alloc(256 * 128);
  float* h2p  = alloc(256 * 128);
  float* structural = alloc(256 * 128);
  float* enc  = alloc(256 * 128);
  float* q    = alloc(256 * 128);
  float* rh1  = alloc(256 * 256);
  float* rh2  = alloc(256 * 256);
  float* rr   = alloc(256 * 128);
  float* oh1  = alloc(256 * 256);
  float* oh2  = alloc(256 * 256);
  float* d1   = alloc(256 * 256);
  float* d2   = alloc(256 * 256);
  float* d3   = alloc(256 * 256);
  float* d4   = alloc(256 * 256);
  float* recon = alloc(256 * 512);
  float* sse  = alloc(256);
  float* vh   = alloc(256 * 256);
  float* vh2  = alloc(256 * 256);
  float* pvar = alloc(256 * 128);
  float* infs = alloc(256 * 128);
  float* joint = alloc(256 * 256);
  float* keys = alloc(256 * 128);
  float* vals = alloc(256 * 128);
  float* hp   = alloc(256 * 3);
  float* mh1  = alloc(256 * 256);
  float* mh2  = alloc(256 * 256);
  float* mout = alloc(256 * 128);
  float* dob  = alloc(256 * 128);
  float* dh2m = alloc(256 * 256);
  float* dh1m = alloc(256 * 256);
  float* wu   = alloc(256);
  float* wa   = alloc(256);
  float* fpb  = alloc(8);
  (void)ws_size; (void)in_sizes; (void)n_in; (void)out_size;

  zero_loss_k<<<1, 32, 0, stream>>>(out);

  // --- path integration MLP -> per-step transition matrices ---
  launch_gemm(stream, actions, 32, pi_w1, 128, pi_b1, nullptr, h1p, 256, 128, 32, FLAG_BIAS | FLAG_RELU);
  launch_gemm(stream, h1p, 128, pi_w2, 128, pi_b2, nullptr, h2p, 256, 128, 128, FLAG_BIAS | FLAG_RELU);
  launch_gemm(stream, h2p, 128, pi_w3, 16384, pi_b3, nullptr, trans, 256, 16384, 128, FLAG_BIAS);
  rnn_scan_k<<<1, 256, 0, stream>>>(trans, rnn_init, structural);

  launch_gemm(stream, sensory, 512, enc_w, 128, nullptr, nullptr, enc, 256, 128, 512, 0);

  // --- retrieve(s, e) chain: q -> meta mlp -> decoder mlp ---
  auto retrieve = [&](const float* sP, int sLda, const float* eP, int eLda, float* dDst) {
    if (sP) {
      launch_gemm(stream, sP, sLda, wq, 128, nullptr, nullptr, q, 256, 128, 128, 0);
      if (eP)
        launch_gemm(stream, eP, eLda, wq + 128 * 128, 128, nullptr, nullptr, q, 256, 128, 128, FLAG_ACCUM);
    } else {
      launch_gemm(stream, eP, eLda, wq + 128 * 128, 128, nullptr, nullptr, q, 256, 128, 128, 0);
    }
    launch_gemm(stream, q, 128, m_w1, 256, m_b1, nullptr, rh1, 256, 256, 128, FLAG_BIAS | FLAG_RELU);
    launch_gemm(stream, rh1, 256, m_w2, 256, m_b2, nullptr, rh2, 256, 256, 256, FLAG_BIAS | FLAG_RELU);
    launch_gemm(stream, rh2, 256, m_w3, 128, m_b3, nullptr, rr, 256, 128, 256, FLAG_BIAS);
    launch_gemm(stream, rr, 128, o_w1, 256, o_b1, nullptr, oh1, 256, 256, 128, FLAG_BIAS | FLAG_RELU);
    launch_gemm(stream, oh1, 256, o_w2, 256, o_b2, nullptr, oh2, 256, 256, 256, FLAG_BIAS | FLAG_RELU);
    launch_gemm(stream, oh2, 256, o_w3, 256, o_b3, nullptr, dDst, 256, 256, 256, FLAG_BIAS);
  };

  // retrieve 1: (structural, 0) -> d1; gen_loss
  retrieve(structural, 128, nullptr, 0, d1);
  launch_gemm(stream, d1 + 128, 256, dec_w, 512, nullptr, nullptr, recon, 256, 512, 128, 0);
  mse_acc_k<<<128, 256, 0, stream>>>(sensory, 512, recon, 512, 256, 9, 1.f / 131072.f, out);

  // retrieve 2: (0, enc) -> d2; relational = 2*mse
  retrieve(nullptr, 0, enc, 128, d2);
  mse_acc_k<<<128, 256, 0, stream>>>(d2, 256, structural, 128, 256, 7, 2.f / 32768.f, out);

  // retrieve 3: (dec_gen_s, enc) -> d3
  retrieve(d1, 256, enc, 128, d3);

  // prediction-variance MLP (input = [corr_s, dec_gen_s, sse], 257 dims split)
  sse_k<<<1, 256, 0, stream>>>(d3, enc, sse);
  launch_gemm(stream, d3, 256, v_w1, 256, nullptr, nullptr, vh, 256, 256, 128, 0);
  launch_gemm(stream, d1, 256, v_w1 + 128 * 256, 256, nullptr, nullptr, vh, 256, 256, 128, FLAG_ACCUM);
  vh_epi_k<<<256, 256, 0, stream>>>(vh, sse, v_w1 + 256 * 256, v_b1);
  launch_gemm(stream, vh, 256, v_w2, 256, v_b2, nullptr, vh2, 256, 256, 256, FLAG_BIAS | FLAG_RELU);
  launch_gemm(stream, vh2, 256, v_w3, 128, v_b3, nullptr, pvar, 256, 128, 256, FLAG_BIAS);

  infs_k<<<128, 256, 0, stream>>>(d1, d3, pvar, ir, infs);
  mse_acc_k<<<128, 256, 0, stream>>>(d1, 256, infs, 128, 256, 7, 1.f / 32768.f, out); // consistency

  // retrieve 4: (inf_s, 0) -> d4; inf_loss
  retrieve(infs, 128, nullptr, 0, d4);
  launch_gemm(stream, d4 + 128, 256, dec_w, 512, nullptr, nullptr, recon, 256, 512, 128, 0);
  mse_acc_k<<<128, 256, 0, stream>>>(sensory, 512, recon, 512, 256, 9, 1.f / 131072.f, out);

  // --- TTT memory write ---
  joint_k<<<256, 256, 0, stream>>>(d4, enc, joint);
  launch_gemm(stream, joint, 256, wk, 128, nullptr, nullptr, keys, 256, 128, 256, 0);
  launch_gemm(stream, joint, 256, wv, 128, nullptr, nullptr, vals, 256, 128, 256, 0);
  hp_k<<<1, 256, 0, stream>>>(joint, w_hp, hp);

  // meta-MLP forward on keys
  launch_gemm(stream, keys, 128, m_w1, 256, m_b1, nullptr, mh1, 256, 256, 128, FLAG_BIAS | FLAG_RELU);
  launch_gemm(stream, mh1, 256, m_w2, 256, m_b2, nullptr, mh2, 256, 256, 256, FLAG_BIAS | FLAG_RELU);
  launch_gemm(stream, mh2, 256, m_w3, 128, m_b3, nullptr, mout, 256, 128, 256, FLAG_BIAS);

  // backward: dout, dh2 = dout@W3^T * (h2>0), dh1 = dh2@W2^T * (h1>0)
  ew_dout_k<<<128, 256, 0, stream>>>(mout, vals, dob, 256 * 128);
  launch_gemm(stream, dob, 128, m_w3, 128, nullptr, mh2, dh2m, 256, 256, 128, FLAG_TRANSB | FLAG_MASK);
  launch_gemm(stream, dh2m, 256, m_w2, 256, nullptr, mh1, dh1m, 256, 256, 256, FLAG_TRANSB | FLAG_MASK);

  coeff_k<<<1, 32, 0, stream>>>(hp, wu, wa, fpb);

  // rank-32 weighted outer-product GEMMs -> final scan states, straight to d_out
  ttt_gemm_k<<<dim3(16, 8, 8),  32, 0, stream>>>(keys, dh1m, m_w1, wu, wa, fpb, out + 1, 128, 256, OFF_W1);
  ttt_gemm_k<<<dim3(16, 16, 8), 32, 0, stream>>>(mh1,  dh2m, m_w2, wu, wa, fpb, out + 1, 256, 256, OFF_W2);
  ttt_gemm_k<<<dim3(8, 16, 8),  32, 0, stream>>>(mh2,  dob,  m_w3, wu, wa, fpb, out + 1, 256, 128, OFF_W3);
  ttt_bias_k<<<dim3(4, 3, 8), 64, 0, stream>>>(dh1m, dh2m, dob, m_b1, m_b2, m_b3, wu, wa, fpb, out + 1);
}
